// AR1ScanTV_90185723281794
// MI455X (gfx1250) — matrix operand order repaired and checked by reading, not compiled, not run on hardware
//
#include <hip/hip_runtime.h>
#include <hip/hip_bf16.h>
#include <math.h>

// ---------------- problem constants (from reference) ----------------
static constexpr int BQ   = 4;
static constexpr int TT   = 4096;
static constexpr int DIN  = 1024;
static constexpr int DH   = 1024;
static constexpr int DOUT = 1024;
static constexpr int MTOT = BQ * TT;          // 16384 rows
static constexpr int CH   = 32;               // scan chunks per batch
static constexpr int LCH  = TT / CH;          // 128 timesteps per chunk

// ---------------- types ----------------
typedef __attribute__((ext_vector_type(16))) __bf16 v16bf;
typedef __attribute__((ext_vector_type(8)))  __bf16 v8bf;
typedef __attribute__((ext_vector_type(8)))  float  v8f;

__device__ __forceinline__ unsigned short f32_to_bf16_rne(float f) {
    unsigned int u = __float_as_uint(f);
    unsigned int r = u + 0x7FFFu + ((u >> 16) & 1u);
    return (unsigned short)(r >> 16);
}

// Generic pointers to LDS are {shared_aperture_hi32 , lds_offset_lo32} (ISA
// 10.2 aperture rules): low 32 bits are the wave-relative LDS byte address.
__device__ __forceinline__ unsigned lds_off_u32(const void* p) {
    return (unsigned)(unsigned long long)p;
}

// CDNA5 async copy: global -> LDS, no VGPR staging, tracked by ASYNCcnt.
__device__ __forceinline__ void async_cp_b128(unsigned lds_dst, const void* gsrc) {
    asm volatile("global_load_async_to_lds_b128 %0, %1, off"
                 :: "v"(lds_dst), "v"((unsigned long long)gsrc)
                 : "memory");
}

// ================== conversion / transpose kernels ==================
__global__ __launch_bounds__(256) void cvt_x_bf16(const float* __restrict__ x,
                                                  unsigned short* __restrict__ xb,
                                                  size_t n4) {
    size_t q = (size_t)blockIdx.x * blockDim.x + threadIdx.x;
    if (q >= n4) return;
    size_t i = q * 4;
    float4 v = *(const float4*)(x + i);
    union { unsigned short u16[4]; uint2 u; } r;
    r.u16[0] = f32_to_bf16_rne(v.x); r.u16[1] = f32_to_bf16_rne(v.y);
    r.u16[2] = f32_to_bf16_rne(v.z); r.u16[3] = f32_to_bf16_rne(v.w);
    *(uint2*)(xb + i) = r.u;
}

// W_ab (DIN x 1+DH) -> Wa[DIN] (f32 col 0) + Wbt (DH x DIN bf16, transposed)
__global__ __launch_bounds__(256) void cvt_wab(const float* __restrict__ Wab,
                                               float* __restrict__ Wa,
                                               unsigned short* __restrict__ Wbt) {
    int idx = blockIdx.x * blockDim.x + threadIdx.x;
    const int total = DIN * (1 + DH);
    if (idx >= total) return;
    int k = idx / (1 + DH);
    int j = idx - k * (1 + DH);
    float v = Wab[idx];
    if (j == 0) Wa[k] = v;
    else        Wbt[(size_t)(j - 1) * DIN + k] = f32_to_bf16_rne(v);
}

// Wy (DH x DOUT) -> Wyt (DOUT x DH bf16, transposed)
__global__ __launch_bounds__(256) void cvt_wy(const float* __restrict__ Wy,
                                              unsigned short* __restrict__ Wyt) {
    int idx = blockIdx.x * blockDim.x + threadIdx.x;
    if (idx >= DH * DOUT) return;
    int k = idx / DOUT;
    int o = idx - k * DOUT;
    Wyt[(size_t)o * DH + k] = f32_to_bf16_rne(Wy[idx]);
}

// ================== a_t = tanh(x . W_ab[:,0] + b_ab[0]) ==================
__global__ __launch_bounds__(256) void gemv_a(const float* __restrict__ x,
                                              const float* __restrict__ Wa,
                                              const float* __restrict__ b_ab,
                                              float* __restrict__ a) {
    int wave = threadIdx.x >> 5;
    int lane = threadIdx.x & 31;
    int row  = blockIdx.x * 8 + wave;          // 8 rows per block
    const float* xr = x + (size_t)row * DIN;
    float s = 0.f;
    #pragma unroll 4
    for (int k = lane; k < DIN; k += 32) s += xr[k] * Wa[k];
    #pragma unroll
    for (int off = 16; off > 0; off >>= 1) s += __shfl_xor(s, off, 32);
    if (lane == 0) a[row] = tanhf(s + b_ab[0]);
}

// ================== bf16 WMMA GEMM: C[M,N] = A[M,K] * Bt[N,K]^T + bias ==================
// Block tile 128x256, 8 waves (2x4), wave tile 64x64 (16 WMMA accumulators).
// Double-buffered LDS fed by global_load_async_to_lds_b128 (ASYNCcnt).
static constexpr int BM = 128, BN = 256, BK = 32;
static constexpr int LSTR = 40;   // LDS row stride (elements): 80B, conflict-free
static constexpr unsigned SZA = BM * LSTR * 2;   // bytes per A buffer
static constexpr unsigned SZB = BN * LSTR * 2;   // bytes per B buffer

__device__ __forceinline__ v16bf frag_A(const unsigned short* base, int khalf) {
    // ISA 16-bit A layout: lane<16 -> K{0..7,16..23}; lane>=16 -> K{8..15,24..31}
    v16bf r;
    ((v8bf*)&r)[0] = *(const v8bf*)(base + khalf * 8);
    ((v8bf*)&r)[1] = *(const v8bf*)(base + 16 + khalf * 8);
    return r;
}
__device__ __forceinline__ v16bf frag_B(const unsigned short* base, int khalf) {
    // B layout (matches A): lane<16 -> K{0..7,16..23}; lane>=16 -> K{8..15,24..31}
    v16bf r;
    ((v8bf*)&r)[0] = *(const v8bf*)(base + khalf * 8);
    ((v8bf*)&r)[1] = *(const v8bf*)(base + 16 + khalf * 8);
    return r;
}

__global__ __launch_bounds__(256) void gemm_bf16(const unsigned short* __restrict__ A,
                                                 const unsigned short* __restrict__ Bt,
                                                 const float* __restrict__ bias,
                                                 float* __restrict__ C,
                                                 int M, int N, int K) {
    __shared__ __align__(16) unsigned short sA[2][BM * LSTR];
    __shared__ __align__(16) unsigned short sB[2][BN * LSTR];

    const int tid    = threadIdx.x;
    const int lane   = tid & 31;
    const int wave   = tid >> 5;
    const int laneLo = lane & 15;
    const int khalf  = lane >> 4;

    const int blockN = blockIdx.x * BN;
    const int blockM = blockIdx.y * BM;
    const int waveM  = (wave >> 2) * 64;   // 0 or 64
    const int waveN  = (wave & 3) * 64;    // 0,64,128,192

    // ---- staging assignment: 16B chunks; A: 512 chunks (2/thread), B: 1024 (4/thread)
    const unsigned short* gA[2];
    unsigned ldsA[2];
    #pragma unroll
    for (int i = 0; i < 2; ++i) {
        int idx = tid + i * 256;            // 0..511
        int row = idx >> 2, c16 = idx & 3;
        gA[i]   = A + (size_t)(blockM + row) * K + c16 * 8;
        ldsA[i] = lds_off_u32(&sA[0][row * LSTR + c16 * 8]);
    }
    const unsigned short* gB[4];
    unsigned ldsB[4];
    #pragma unroll
    for (int i = 0; i < 4; ++i) {
        int idx = tid + i * 256;            // 0..1023
        int row = idx >> 2, c16 = idx & 3;
        gB[i]   = Bt + (size_t)(blockN + row) * K + c16 * 8;
        ldsB[i] = lds_off_u32(&sB[0][row * LSTR + c16 * 8]);
    }

    v8f acc[4][4];
    #pragma unroll
    for (int mi = 0; mi < 4; ++mi)
        #pragma unroll
        for (int ni = 0; ni < 4; ++ni)
            acc[mi][ni] = (v8f)(0.0f);

    const int NK = K / BK;

    // prologue: stage tile 0 into buffer 0
    #pragma unroll
    for (int i = 0; i < 2; ++i) async_cp_b128(ldsA[i], gA[i]);
    #pragma unroll
    for (int i = 0; i < 4; ++i) async_cp_b128(ldsB[i], gB[i]);

    for (int kk = 0; kk < NK; ++kk) {
        const int buf = kk & 1;
        if (kk + 1 < NK) {
            const int k1 = (kk + 1) * BK;
            const unsigned boffA = (buf ^ 1) ? SZA : 0u;
            const unsigned boffB = (buf ^ 1) ? SZB : 0u;
            #pragma unroll
            for (int i = 0; i < 2; ++i) async_cp_b128(ldsA[i] + boffA, gA[i] + k1);
            #pragma unroll
            for (int i = 0; i < 4; ++i) async_cp_b128(ldsB[i] + boffB, gB[i] + k1);
            // 6 outstanding from the stage just issued; <=6 means tile kk is resident
            asm volatile("s_wait_asynccnt 0x6" ::: "memory");
        } else {
            asm volatile("s_wait_asynccnt 0x0" ::: "memory");
        }
        __syncthreads();   // all waves' async writes for tile kk visible

        v16bf afr[4], bfr[4];
        #pragma unroll
        for (int mi = 0; mi < 4; ++mi)
            afr[mi] = frag_A(&sA[buf][(waveM + mi * 16 + laneLo) * LSTR], khalf);
        #pragma unroll
        for (int ni = 0; ni < 4; ++ni)
            bfr[ni] = frag_B(&sB[buf][(waveN + ni * 16 + laneLo) * LSTR], khalf);

        #pragma unroll
        for (int mi = 0; mi < 4; ++mi)
            #pragma unroll
            for (int ni = 0; ni < 4; ++ni)
                acc[mi][ni] = __builtin_amdgcn_wmma_f32_16x16x32_bf16(
                    false, afr[mi], false, bfr[ni], (short)0, acc[mi][ni], false, false);

        __syncthreads();   // done reading buf before it is overwritten at kk+1
    }

    // epilogue: C layout VGPR r -> (M=r, N=lane) lanes 0-15 ; (M=r+8, N=lane-16) lanes 16-31
    #pragma unroll
    for (int ni = 0; ni < 4; ++ni) {
        int n = blockN + waveN + ni * 16 + laneLo;
        float bv = bias[n];
        #pragma unroll
        for (int mi = 0; mi < 4; ++mi) {
            #pragma unroll
            for (int r = 0; r < 8; ++r) {
                int m = blockM + waveM + mi * 16 + r + khalf * 8;
                C[(size_t)m * N + n] = acc[mi][ni][r] + bv;
            }
        }
    }
}

// ================== chunked AR(1) scan ==================
// Phase 1: per (batch, chunk): local scan in-place over bh, record inclusive
// prefix products p_t and chunk-end carry.
__global__ __launch_bounds__(256) void scan_phase1(const float* __restrict__ a,
                                                   float* __restrict__ bh,
                                                   float* __restrict__ p,
                                                   float* __restrict__ cend) {
    int b = blockIdx.y;
    int c = blockIdx.x;
    int col = threadIdx.x * 4;
    size_t rowBase = (size_t)b * TT + (size_t)c * LCH;
    float4 h = make_float4(0.f, 0.f, 0.f, 0.f);
    float pp = 1.0f;
    for (int t = 0; t < LCH; ++t) {
        size_t row = rowBase + t;
        float at = a[row];
        pp *= at;
        float4* ptr = (float4*)(bh + row * DH + col);
        float4 bt = *ptr;
        h.x = fmaf(at, h.x, bt.x);
        h.y = fmaf(at, h.y, bt.y);
        h.z = fmaf(at, h.z, bt.z);
        h.w = fmaf(at, h.w, bt.w);
        *ptr = h;
        if (threadIdx.x == 0) p[row] = pp;
    }
    *(float4*)(cend + ((size_t)(b * CH + c)) * DH + col) = h;
}

// Phase 2: per (b, j): exclusive scan of chunk carries.
__global__ __launch_bounds__(256) void scan_phase2(const float* __restrict__ p,
                                                   const float* __restrict__ cend,
                                                   float* __restrict__ cin) {
    int idx = blockIdx.x * blockDim.x + threadIdx.x;   // 0 .. B*DH-1
    int b = idx / DH;
    int j = idx - b * DH;
    float H = 0.f;
    for (int c = 0; c < CH; ++c) {
        size_t off = ((size_t)(b * CH + c)) * DH + j;
        cin[off] = H;
        float Ac = p[(size_t)b * TT + (size_t)c * LCH + (LCH - 1)];
        H = fmaf(Ac, H, cend[off]);
    }
}

// Phase 3: h = h_local + p_t * carry_in, fused with bf16 conversion for GEMM2.
__global__ __launch_bounds__(256) void scan_fixup_cvt(const float* __restrict__ hloc,
                                                      const float* __restrict__ p,
                                                      const float* __restrict__ cin,
                                                      unsigned short* __restrict__ hb) {
    size_t q = (size_t)blockIdx.x * blockDim.x + threadIdx.x;   // quad index
    size_t row = q >> 8;                 // DH/4 = 256 quads per row
    int col = (int)(q & 255) * 4;
    int b  = (int)(row / TT);
    int tt = (int)(row - (size_t)b * TT);
    int c  = tt / LCH;
    float pt = p[row];
    float4 ci = *(const float4*)(cin + ((size_t)(b * CH + c)) * DH + col);
    float4 hl = *(const float4*)(hloc + row * DH + col);
    union { unsigned short u16[4]; uint2 u; } r;
    r.u16[0] = f32_to_bf16_rne(fmaf(pt, ci.x, hl.x));
    r.u16[1] = f32_to_bf16_rne(fmaf(pt, ci.y, hl.y));
    r.u16[2] = f32_to_bf16_rne(fmaf(pt, ci.z, hl.z));
    r.u16[3] = f32_to_bf16_rne(fmaf(pt, ci.w, hl.w));
    *(uint2*)(hb + row * DH + col) = r.u;
}

// ================== host launcher ==================
extern "C" void kernel_launch(void* const* d_in, const int* in_sizes, int n_in,
                              void* d_out, int out_size, void* d_ws, size_t ws_size,
                              hipStream_t stream) {
    const float* x    = (const float*)d_in[0];   // (B,T,DIN)
    const float* Wab  = (const float*)d_in[1];   // (DIN, 1+DH)
    const float* b_ab = (const float*)d_in[2];   // (1+DH,)
    const float* Wy   = (const float*)d_in[3];   // (DH, DOUT)
    const float* by   = (const float*)d_in[4];   // (DOUT,)
    float* out = (float*)d_out;

    // workspace carve-out (256B aligned slices)
    char* w = (char*)d_ws;
    auto carve = [&](size_t bytes) {
        char* r = w;
        w += (bytes + 255) & ~(size_t)255;
        return r;
    };
    unsigned short* xb   = (unsigned short*)carve((size_t)MTOT * DIN * 2);  // 32 MB
    unsigned short* wbt  = (unsigned short*)carve((size_t)DH * DIN * 2);    //  2 MB
    unsigned short* wyt  = (unsigned short*)carve((size_t)DOUT * DH * 2);   //  2 MB
    float* wa   = (float*)carve((size_t)DIN * 4);
    float* a    = (float*)carve((size_t)MTOT * 4);
    float* p    = (float*)carve((size_t)MTOT * 4);
    float* bh   = (float*)carve((size_t)MTOT * DH * 4);                     // 64 MB (b then h_local)
    float* cend = (float*)carve((size_t)BQ * CH * DH * 4);
    float* cin  = (float*)carve((size_t)BQ * CH * DH * 4);
    unsigned short* hb = (unsigned short*)carve((size_t)MTOT * DH * 2);     // 32 MB
    (void)ws_size; (void)n_in; (void)in_sizes; (void)out_size;

    // 1) conversions / transposes
    {
        size_t n4 = (size_t)MTOT * DIN / 4;
        cvt_x_bf16<<<(unsigned)((n4 + 255) / 256), 256, 0, stream>>>(x, xb, n4);
        int totW = DIN * (1 + DH);
        cvt_wab<<<(totW + 255) / 256, 256, 0, stream>>>(Wab, wa, wbt);
        cvt_wy<<<(DH * DOUT + 255) / 256, 256, 0, stream>>>(Wy, wyt);
    }
    // 2) a_t = tanh(x . Wab[:,0] + b_ab[0])
    gemv_a<<<MTOT / 8, 256, 0, stream>>>(x, wa, b_ab, a);

    // 3) b_t = x @ Wab[:,1:] + b_ab[1:]   (bf16 WMMA, async double-buffered)
    {
        dim3 grid(DH / BN, MTOT / BM);
        gemm_bf16<<<grid, 256, 0, stream>>>(xb, wbt, b_ab + 1, bh, MTOT, DH, DIN);
    }
    // 4) chunked AR(1) scan
    {
        dim3 g1(CH, BQ);
        scan_phase1<<<g1, 256, 0, stream>>>(a, bh, p, cend);
        scan_phase2<<<(BQ * DH) / 256, 256, 0, stream>>>(p, cend, cin);
        size_t nq = (size_t)MTOT * DH / 4;
        scan_fixup_cvt<<<(unsigned)(nq / 256), 256, 0, stream>>>(bh, p, cin, hb);
    }
    // 5) y = h @ Wy + by   (bf16 WMMA, async double-buffered)
    {
        dim3 grid(DOUT / BN, MTOT / BM);
        gemm_bf16<<<grid, 256, 0, stream>>>(hb, wyt, by, out, MTOT, DOUT, DH);
    }
}
